// RelaxedContrastiveLoss_25108378812764
// MI455X (gfx1250) — matrix operand order repaired and checked by reading, not compiled, not run on hardware
//
#include <hip/hip_runtime.h>
#include <hip/hip_bf16.h>
#include <math.h>

// Problem constants (match reference)
#define BN      8192
#define DK      256
#define INV_T   20.0f          // 1 / TEMPERATURE
#define LAMBDA  0.7f
#define NSLICE  8              // column slices
#define CT_TOT  (BN / 16)      // 512 column tiles
#define CT_PER  (CT_TOT / NSLICE)

typedef __attribute__((ext_vector_type(16))) _Float16 v16h;
typedef __attribute__((ext_vector_type(8)))  _Float16 v8h;
typedef __attribute__((ext_vector_type(8)))  float    v8f;

// ---------------------------------------------------------------- zero scratch
__global__ void k_zero(float* __restrict__ p, int n) {
    int i = blockIdx.x * blockDim.x + threadIdx.x;
    if (i < n) p[i] = 0.0f;
}

// ------------------------------------------------- row-normalize + f32 -> f16
// one wave per row; lane handles elements lane + 32*i (coalesced)
__global__ void k_normalize(const float* __restrict__ feat,
                            _Float16* __restrict__ fhat) {
    int wave = (blockIdx.x * blockDim.x + threadIdx.x) >> 5;
    int lane = threadIdx.x & 31;
    const float* row = feat + (size_t)wave * DK;
    float vals[8];
    float s = 0.0f;
#pragma unroll
    for (int i = 0; i < 8; ++i) {
        float x = row[lane + 32 * i];
        vals[i] = x;
        s += x * x;
    }
#pragma unroll
    for (int m = 16; m >= 1; m >>= 1) s += __shfl_xor(s, m, 32);
    float inv = rsqrtf(s);
    _Float16* orow = fhat + (size_t)wave * DK;
#pragma unroll
    for (int i = 0; i < 8; ++i)
        orow[lane + 32 * i] = (_Float16)(vals[i] * inv);
}

// ------------------------------------- fused WMMA GEMM + rowwise reductions
// Block = 8 waves sharing ONE column slice, each wave owning its own row tile
// -> all waves request identical B tiles (WGP$ hits), A is hoisted per wave.
// Two column tiles per iteration so WMMA co-executes with the exp epilogue.
__global__ void __launch_bounds__(256)
k_sim(const _Float16* __restrict__ fhat,
      const int* __restrict__ labels,
      float* __restrict__ denom,  float* __restrict__ posCnt,
      float* __restrict__ posSim, float* __restrict__ highSum,
      float* __restrict__ highCnt) {
    int w      = threadIdx.x >> 5;            // wave in block: 0..7
    int lane   = threadIdx.x & 31;
    int laneHi = lane >> 4;                   // 0: lanes 0-15, 1: lanes 16-31
    int lan16  = lane & 15;
    int rowTile = ((blockIdx.x & 63) << 3) | w;   // 512 row tiles
    int slice   = blockIdx.x >> 6;                // 8 column slices
    int row0    = rowTile * 16;

    // ---- hoist A fragments (invariant over the column sweep) --------------
    // A 16x32 f16 layout: lane m holds row m; lo-half lanes K={0..7,16..23},
    // hi-half lanes K={8..15,24..31}  (ISA 7.12.2)
    v16h afrag[8];
    {
        const _Float16* ap =
            fhat + (size_t)(row0 + lan16) * DK + (laneHi ? 8 : 0);
#pragma unroll
        for (int ks = 0; ks < 8; ++ks) {
            v8h lo = *(const v8h*)(ap + ks * 32);
            v8h hi = *(const v8h*)(ap + ks * 32 + 16);
            afrag[ks] = __builtin_shufflevector(
                lo, hi, 0,1,2,3,4,5,6,7,8,9,10,11,12,13,14,15);
        }
    }
    int labRow[8];
#pragma unroll
    for (int v = 0; v < 8; ++v)
        labRow[v] = labels[row0 + v + laneHi * 8];

    float denomA[8], pcA[8], psA[8], hsA[8], hcA[8];
#pragma unroll
    for (int v = 0; v < 8; ++v) {
        denomA[v] = 0.f; pcA[v] = 0.f; psA[v] = 0.f; hsA[v] = 0.f; hcA[v] = 0.f;
    }

    int ct0 = slice * CT_PER;
    for (int ct = ct0; ct < ct0 + CT_PER; ct += 2) {
        // ---- two independent column tiles per iteration -------------------
        int colB0 = ct * 16 + lan16;
        int colB1 = colB0 + 16;
        int labC0 = labels[colB0];
        int labC1 = labels[colB1];
        // B 32x16 f16 layout: lane n holds column n; lo-half lanes K={0..7,16..23},
        // hi-half lanes K={8..15,24..31}
        const _Float16* bp0 =
            fhat + (size_t)colB0 * DK + (laneHi ? 8 : 0);
        const _Float16* bp1 =
            fhat + (size_t)colB1 * DK + (laneHi ? 8 : 0);

        // prefetch next iteration's column rows into WGP$ (counter-free)
        __builtin_prefetch(fhat + (size_t)(colB0 + 32) * DK, 0, 3);
        __builtin_prefetch(fhat + (size_t)(colB1 + 32) * DK, 0, 3);

        v8f c0 = {};
        v8f c1 = {};
#pragma unroll
        for (int ks = 0; ks < 8; ++ks) {
            v8h lo0 = *(const v8h*)(bp0 + ks * 32);
            v8h hi0 = *(const v8h*)(bp0 + ks * 32 + 16);
            v16h b0 = __builtin_shufflevector(
                lo0, hi0, 0,1,2,3,4,5,6,7,8,9,10,11,12,13,14,15);
            c0 = __builtin_amdgcn_wmma_f32_16x16x32_f16(
                     false, afrag[ks], false, b0, (short)0, c0, false, false);
            v8h lo1 = *(const v8h*)(bp1 + ks * 32);
            v8h hi1 = *(const v8h*)(bp1 + ks * 32 + 16);
            v16h b1 = __builtin_shufflevector(
                lo1, hi1, 0,1,2,3,4,5,6,7,8,9,10,11,12,13,14,15);
            c1 = __builtin_amdgcn_wmma_f32_16x16x32_f16(
                     false, afrag[ks], false, b1, (short)0, c1, false, false);
        }

        // ---- branchless epilogue: D layout VGPR v -> row row0+v+8*laneHi --
#pragma unroll
        for (int v = 0; v < 8; ++v) {
            int rr = row0 + v + laneHi * 8;
            {
                float dot  = c0[v];
                float sim  = dot * INV_T;
                float e    = __expf(sim);
                float nd   = (rr != colB0)        ? 1.0f : 0.0f;
                float pos  = (labRow[v] == labC0) ? nd   : 0.0f;
                float high = (dot > LAMBDA)       ? pos  : 0.0f;
                denomA[v] = fmaf(nd,  e,   denomA[v]);
                pcA[v]   += pos;
                psA[v]    = fmaf(pos, sim, psA[v]);
                hsA[v]    = fmaf(high, e,  hsA[v]);
                hcA[v]   += high;
            }
            {
                float dot  = c1[v];
                float sim  = dot * INV_T;
                float e    = __expf(sim);
                float nd   = (rr != colB1)        ? 1.0f : 0.0f;
                float pos  = (labRow[v] == labC1) ? nd   : 0.0f;
                float high = (dot > LAMBDA)       ? pos  : 0.0f;
                denomA[v] = fmaf(nd,  e,   denomA[v]);
                pcA[v]   += pos;
                psA[v]    = fmaf(pos, sim, psA[v]);
                hsA[v]    = fmaf(high, e,  hsA[v]);
                hcA[v]   += high;
            }
        }
    }

    // ---- reduce the 16 lanes that share each row, then merge globally -----
#pragma unroll
    for (int v = 0; v < 8; ++v) {
        float d = denomA[v], pc = pcA[v], ps = psA[v], hs = hsA[v], hc = hcA[v];
#pragma unroll
        for (int m = 8; m >= 1; m >>= 1) {
            d  += __shfl_xor(d,  m, 16);
            pc += __shfl_xor(pc, m, 16);
            ps += __shfl_xor(ps, m, 16);
            hs += __shfl_xor(hs, m, 16);
            hc += __shfl_xor(hc, m, 16);
        }
        if (lan16 == 0) {
            int rr = row0 + v + laneHi * 8;
            atomicAdd(&denom[rr],   d);
            atomicAdd(&posCnt[rr],  pc);
            atomicAdd(&posSim[rr],  ps);
            atomicAdd(&highSum[rr], hs);
            atomicAdd(&highCnt[rr], hc);
        }
    }
}

// --------------------------------------------- per-row logs -> 4 scalar sums
__global__ void k_rows(const float* __restrict__ denom,
                       const float* __restrict__ posCnt,
                       const float* __restrict__ posSim,
                       const float* __restrict__ highSum,
                       const float* __restrict__ highCnt,
                       float* __restrict__ S) {
    int i    = blockIdx.x * blockDim.x + threadIdx.x;
    int lane = threadIdx.x & 31;
    float scl = 0.f, pc = 0.f, rl = 0.f, hc = 0.f;
    if (i < BN) {
        pc  = posCnt[i];
        scl = pc * logf(denom[i] + 1e-8f) - posSim[i];
        float hs = highSum[i];
        const float SELF = __expf(INV_T);      // exp(1/T)
        rl = (hs > 0.f) ? logf(hs + SELF) : 0.f;
        hc = highCnt[i];
    }
#pragma unroll
    for (int m = 16; m >= 1; m >>= 1) {
        scl += __shfl_xor(scl, m, 32);
        pc  += __shfl_xor(pc,  m, 32);
        rl  += __shfl_xor(rl,  m, 32);
        hc  += __shfl_xor(hc,  m, 32);
    }
    if (lane == 0) {
        atomicAdd(&S[0], scl);
        atomicAdd(&S[1], pc);
        atomicAdd(&S[2], rl);
        atomicAdd(&S[3], hc);
    }
}

// ---------------------------------------------------------------- final scalar
__global__ void k_final(const float* __restrict__ S, float* __restrict__ out) {
    float scl = (S[1] > 0.f) ? S[0] / S[1] : 0.f;
    float rlx = (S[3] > 0.f) ? S[2] / S[3] : 0.f;
    out[0] = scl + 1.0f * rlx;                 // BETA = 1.0
}

// =============================================================== entry point
extern "C" void kernel_launch(void* const* d_in, const int* in_sizes, int n_in,
                              void* d_out, int out_size, void* d_ws, size_t ws_size,
                              hipStream_t stream) {
    const float* feat   = (const float*)d_in[0];
    const int*   labels = (const int*)d_in[1];
    float*       out    = (float*)d_out;

    char*     ws      = (char*)d_ws;
    _Float16* fhat    = (_Float16*)ws;                       // 8192*256*2 = 4 MB
    float*    denom   = (float*)(ws + (size_t)BN * DK * 2);
    float*    posCnt  = denom   + BN;
    float*    posSim  = posCnt  + BN;
    float*    highSum = posSim  + BN;
    float*    highCnt = highSum + BN;
    float*    S       = highCnt + BN;                        // 8 scalars

    // zero row accumulators + scalars (harness does not re-zero scratch)
    int nz = 5 * BN + 8;
    k_zero<<<(nz + 255) / 256, 256, 0, stream>>>(denom, nz);

    // normalize: one wave per row, 8 waves per block
    k_normalize<<<BN / 8, 256, 0, stream>>>(feat, fhat);

    // fused GEMM + reductions:
    // 512 blocks = 64 row-groups (8 row tiles each, one per wave) x 8 slices
    k_sim<<<(CT_TOT * NSLICE) / 8, 256, 0, stream>>>(
        fhat, labels, denom, posCnt, posSim, highSum, highCnt);

    // per-row epilogue + scalar combine
    k_rows<<<BN / 256, 256, 0, stream>>>(denom, posCnt, posSim, highSum,
                                         highCnt, S);
    k_final<<<1, 1, 0, stream>>>(S, out);
}
